// SelectK_23295902613536
// MI455X (gfx1250) — matrix operand order repaired and checked by reading, not compiled, not run on hardware
//
#include <hip/hip_runtime.h>

// ---------------- types for WMMA fragments (wave32, gfx1250) ----------------
typedef __bf16 bf16_t;
typedef bf16_t v16bf __attribute__((ext_vector_type(16)));
typedef float  v8f   __attribute__((ext_vector_type(8)));
typedef unsigned int u32x4 __attribute__((ext_vector_type(4)));
typedef unsigned int u32x2 __attribute__((ext_vector_type(2)));

union FragU { v16bf v; u32x4 q[2]; };

__device__ __forceinline__ unsigned short f2bfu(float f) {
  unsigned u = __builtin_bit_cast(unsigned, f);
  u += 0x7FFFu + ((u >> 16) & 1u);            // round-to-nearest-even
  return (unsigned short)(u >> 16);
}
__device__ __forceinline__ float bfu2f(unsigned short h) {
  unsigned u = ((unsigned)h) << 16;
  return __builtin_bit_cast(float, u);
}

// ---------------- problem constants ----------------
#define BB   8
#define SS   256
#define SB   2048      /* B*S rows */
#define DD   300
#define DP   320
#define HH   1150
#define HP   1152
#define G3   3450      /* 3H */
#define G3P  3456
#define NG   40000
#define NS   25000
#define RBLK 216       /* persistent blocks in recurrence = G3P/16 */

// ---------------- grid-wide barrier (monotonic counter) ---------------------
__device__ __forceinline__ void gsync(int* cnt, int nb, int& phase) {
  __syncthreads();
  __threadfence();                                  // release
  if (threadIdx.x == 0) {
    atomicAdd(cnt, 1);
    int target = (phase + 1) * nb;
    while (atomicAdd(cnt, 0) < target) __builtin_amdgcn_s_sleep(2);
  }
  phase++;
  __syncthreads();
  __threadfence();                                  // acquire
}

// ---------------- workspace zeroing ----------------
__global__ void zero_ws_kernel(unsigned int* p, long n) {
  long i = (long)blockIdx.x * blockDim.x + threadIdx.x;
  long st = (long)gridDim.x * blockDim.x;
  for (; i < n; i += st) p[i] = 0u;
}

// ---------------- embedding gather: x0[s*8+b][0..299] = bf16(X[tok[b][s]]) --
__global__ void gather_kernel(const int* __restrict__ tok,
                              const float* __restrict__ X,
                              unsigned short* __restrict__ x0) {
  int row = blockIdx.x;                 // row = s*8 + b
  int s = row >> 3, b = row & 7;
  int idx = tok[b * SS + s];
  const float* src = X + (size_t)idx * DD;
  unsigned short* dst = x0 + (size_t)row * DP;
  for (int c = threadIdx.x; c < DD; c += blockDim.x) dst[c] = f2bfu(src[c]);
}

// ---------------- generic GEMM: C[M=2048,N] = A(bf16) @ W(f32,[N,K])^T + bias
// grid.x = ceil(N/32) (32-col tiles), grid.y = 16 (8 M-tiles/block, 1/wave).
// Double-buffered LDS W tile (32 rows x 32 k, bf16); one A-frag feeds 2 WMMAs.
__global__ void wmma_gemm_kernel(const unsigned short* __restrict__ A, int lda,
                                 const float* __restrict__ W, int K, int Kp, int N,
                                 const float* __restrict__ bias,
                                 float* __restrict__ C, int ldc, int Nstore,
                                 int permute) {
  __shared__ unsigned short wt[2][32 * 32];
  int tid = threadIdx.x;
  int ntile = blockIdx.x;
  int wave = tid >> 5;
  int lane = tid & 31;
  int mt = blockIdx.y * 8 + wave;
  int m  = lane & 15, kb  = (lane >> 4) * 8;    // A-frag: row, k-offset
  int nn = lane & 15, kb2 = (lane >> 4) * 8;    // B-frag: col, k-offset
  int fr = tid >> 3, fc = (tid & 7) * 4;        // fill: row 0..31, col {0,4,..28}
  size_t NK2 = (size_t)N * K - 2;               // flat clamp (pair-aligned, even)
  size_t frow = (size_t)(ntile * 32 + fr) * K + fc;
  v8f acc0 = {}, acc1 = {};

  // prime buffer 0 (k0 = 0)
  {
    size_t f0 = frow < NK2 ? frow : NK2;
    size_t f1 = (frow + 2) < NK2 ? (frow + 2) : NK2;
    float2 wa = *(const float2*)(W + f0);
    float2 wb = *(const float2*)(W + f1);
    u32x2 pk = { f2bfu(wa.x) | ((unsigned)f2bfu(wa.y) << 16),
                 f2bfu(wb.x) | ((unsigned)f2bfu(wb.y) << 16) };
    *(u32x2*)&wt[0][fr * 32 + fc] = pk;
  }
  __syncthreads();

  for (int k0 = 0; k0 < Kp; k0 += 32) {
    int cur = (k0 >> 5) & 1;
    if (k0 + 32 < Kp) {                        // prefetch next tile into buf^1
      size_t flat = frow + (size_t)(k0 + 32);
      size_t f0 = flat < NK2 ? flat : NK2;
      size_t f1 = (flat + 2) < NK2 ? (flat + 2) : NK2;
      float2 wa = *(const float2*)(W + f0);
      float2 wb = *(const float2*)(W + f1);
      u32x2 pk = { f2bfu(wa.x) | ((unsigned)f2bfu(wa.y) << 16),
                   f2bfu(wb.x) | ((unsigned)f2bfu(wb.y) << 16) };
      *(u32x2*)&wt[cur ^ 1][fr * 32 + fc] = pk;
    }
    FragU ua, ub0, ub1;
    const unsigned short* ap = A + (size_t)(mt * 16 + m) * lda + k0 + kb;
    ua.q[0] = *(const u32x4*)ap;
    ua.q[1] = *(const u32x4*)(ap + 16);
    const unsigned short* b0 = &wt[cur][nn * 32 + kb2];
    const unsigned short* b1 = &wt[cur][(16 + nn) * 32 + kb2];
    ub0.q[0] = *(const u32x4*)b0;  ub0.q[1] = *(const u32x4*)(b0 + 16);
    ub1.q[0] = *(const u32x4*)b1;  ub1.q[1] = *(const u32x4*)(b1 + 16);
    acc0 = __builtin_amdgcn_wmma_f32_16x16x32_bf16(false, ua.v, false, ub0.v,
                                                   (short)0, acc0, false, false);
    acc1 = __builtin_amdgcn_wmma_f32_16x16x32_bf16(false, ua.v, false, ub1.v,
                                                   (short)0, acc1, false, false);
    __syncthreads();
  }

  // store both 16-col sub-tiles (C layout: lane holds col nn, rows (l>>4)*8+v)
  for (int half = 0; half < 2; ++half) {
    int ncol = ntile * 32 + half * 16 + nn;
    v8f& acc = half ? acc1 : acc0;
    if (ncol < Nstore) {
      float bv = bias[ncol < N ? ncol : N - 1];
      for (int v = 0; v < 8; ++v) {
        int arow = mt * 16 + (lane >> 4) * 8 + v;            // act row = s*8+b
        int orow = permute ? ((arow & 7) * SS + (arow >> 3)) : arow;  // b*256+s
        C[(size_t)orow * ldc + ncol] = acc[v] + bv;
      }
    }
  }
}

// ---------------- persistent GRU recurrence (one launch per layer) ----------
// 216 blocks x 32 threads. Block nt caches w_hh rows [nt*16, nt*16+16) in LDS
// (bf16, 36KB) once; loops 256 steps: hp-tile GEMM -> barrier -> gate update
// -> barrier. h is [16][HP] bf16 (rows 8..15 stay zero).
__global__ void gru_recur_kernel(const float* __restrict__ xp,
                                 const float* __restrict__ w_hh,
                                 const float* __restrict__ b_hh,
                                 unsigned short* __restrict__ h,
                                 float* __restrict__ hp,
                                 unsigned short* __restrict__ out,
                                 int* cnt) {
  __shared__ unsigned short wt[16 * HP];
  int tid = threadIdx.x;                 // 0..31 (one wave)
  int nt = blockIdx.x;                   // N tile 0..215
  int phase = 0;

  // load this block's weight slice into LDS (bf16, zero-padded)
  for (int i = tid; i < 16 * HP; i += 32) {
    int r = i / HP, c = i % HP;
    int gr = nt * 16 + r;
    float f = (gr < G3 && c < HH) ? w_hh[(size_t)gr * HH + c] : 0.f;
    wt[i] = f2bfu(f);
  }
  // cooperative zero of h (h0 = 0; also clears previous layer's state)
  for (int i = nt * 32 + tid; i < 16 * HP; i += RBLK * 32) h[i] = 0;
  gsync(cnt, RBLK, phase);

  int lane = tid;
  int m  = lane & 15, kb  = (lane >> 4) * 8;
  int nn = lane & 15, kb2 = (lane >> 4) * 8;

  for (int t = 0; t < SS; ++t) {
    // ---- hp[:, nt*16 .. +15] = h @ w_hh^T (K = HP, bf16 WMMA, f32 acc) ----
    v8f acc = {};
#pragma unroll 4
    for (int k0 = 0; k0 < HP; k0 += 32) {
      FragU ua, ub;
      const unsigned short* ap = h + (size_t)m * HP + k0 + kb;
      ua.q[0] = *(const u32x4*)ap;
      ua.q[1] = *(const u32x4*)(ap + 16);
      const unsigned short* bp = &wt[nn * HP + k0 + kb2];
      ub.q[0] = *(const u32x4*)bp;
      ub.q[1] = *(const u32x4*)(bp + 16);
      acc = __builtin_amdgcn_wmma_f32_16x16x32_bf16(false, ua.v, false, ub.v,
                                                    (short)0, acc, false, false);
    }
    int ncol = nt * 16 + nn;
    for (int v = 0; v < 8; ++v) {
      int mr = (lane >> 4) * 8 + v;
      hp[(size_t)mr * G3P + ncol] = acc[v];
    }
    gsync(cnt, RBLK, phase);

    // ---- gate update: h_new = (1-z)*n + z*h  (distributed over all blocks)
    for (int i = nt * 32 + tid; i < BB * HH; i += RBLK * 32) {
      int b = i / HH, c = i % HH;
      size_t xrow = (size_t)(t * BB + b) * G3P;
      float xr = xp[xrow + c], xz = xp[xrow + HH + c], xn = xp[xrow + 2 * HH + c];
      size_t hrow = (size_t)b * G3P;
      float hr = hp[hrow + c] + b_hh[c];
      float hz = hp[hrow + HH + c] + b_hh[HH + c];
      float hn = hp[hrow + 2 * HH + c] + b_hh[2 * HH + c];
      float r = 1.f / (1.f + __expf(-(xr + hr)));
      float z = 1.f / (1.f + __expf(-(xz + hz)));
      float ng = tanhf(xn + r * hn);
      float hprev = bfu2f(h[(size_t)b * HP + c]);
      float hnew = (1.f - z) * ng + z * hprev;
      unsigned short hv = f2bfu(hnew);
      h[(size_t)b * HP + c] = hv;
      out[(size_t)(t * BB + b) * HP + c] = hv;
    }
    gsync(cnt, RBLK, phase);
  }
}

// ---------------- in-place row log_softmax ----------------
__global__ void logsoftmax_kernel(float* __restrict__ out, int N) {
  __shared__ float red[256];
  float* row = out + (size_t)blockIdx.x * N;
  int tid = threadIdx.x;
  float mx = -3.4e38f;
  for (int c = tid; c < N; c += 256) mx = fmaxf(mx, row[c]);
  red[tid] = mx; __syncthreads();
  for (int s = 128; s > 0; s >>= 1) {
    if (tid < s) red[tid] = fmaxf(red[tid], red[tid + s]);
    __syncthreads();
  }
  mx = red[0]; __syncthreads();
  float sum = 0.f;
  for (int c = tid; c < N; c += 256) sum += __expf(row[c] - mx);
  red[tid] = sum; __syncthreads();
  for (int s = 128; s > 0; s >>= 1) {
    if (tid < s) red[tid] += red[tid + s];
    __syncthreads();
  }
  float l = __logf(red[0]) + mx;
  for (int c = tid; c < N; c += 256) row[c] = row[c] - l;
}

// ---------------- host-side orchestration ----------------
extern "C" void kernel_launch(void* const* d_in, const int* in_sizes, int n_in,
                              void* d_out, int out_size, void* d_ws, size_t ws_size,
                              hipStream_t stream) {
  (void)in_sizes; (void)n_in; (void)out_size;
  const int*   tok    = (const int*)  d_in[0];
  const float* X      = (const float*)d_in[1];
  const float* w_ih0  = (const float*)d_in[2];
  const float* w_hh0  = (const float*)d_in[3];
  const float* b_ih0  = (const float*)d_in[4];
  const float* b_hh0  = (const float*)d_in[5];
  const float* w_ih1  = (const float*)d_in[6];
  const float* w_hh1  = (const float*)d_in[7];
  const float* b_ih1  = (const float*)d_in[8];
  const float* b_hh1  = (const float*)d_in[9];
  const float* w_ih2  = (const float*)d_in[10];
  const float* w_hh2  = (const float*)d_in[11];
  const float* b_ih2  = (const float*)d_in[12];
  const float* b_hh2  = (const float*)d_in[13];
  const float* w_ihs  = (const float*)d_in[14];
  const float* w_hhs  = (const float*)d_in[15];
  const float* b_ihs  = (const float*)d_in[16];
  const float* b_hhs  = (const float*)d_in[17];
  const float* w_glob = (const float*)d_in[18];
  const float* b_glob = (const float*)d_in[19];
  const float* w_sens = (const float*)d_in[20];
  const float* b_sens = (const float*)d_in[21];

  // workspace layout (256-byte aligned slices)
  char* ws = (char*)d_ws;
  size_t off = 0;
  auto take = [&](size_t bytes) -> void* {
    void* p = ws + off;
    off += (bytes + 255) & ~(size_t)255;
    return p;
  };
  int*            cnt  = (int*)           take(64);
  unsigned short* x0   = (unsigned short*)take((size_t)SB * DP * 2);
  unsigned short* out0 = (unsigned short*)take((size_t)SB * HP * 2);
  unsigned short* out1 = (unsigned short*)take((size_t)SB * HP * 2);
  unsigned short* out2 = (unsigned short*)take((size_t)SB * HP * 2);
  unsigned short* sout = (unsigned short*)take((size_t)SB * HP * 2);
  unsigned short* h    = (unsigned short*)take((size_t)16 * HP * 2);
  float*          hp   = (float*)         take((size_t)16 * G3P * 4);
  float*          xp   = (float*)         take((size_t)SB * G3P * 4);
  if (ws_size < off) return;  // not enough scratch; bail deterministically

  float* outG = (float*)d_out;
  float* outS = outG + (size_t)SB * NG;

  // 1. zero workspace (pads must be 0 so clamped weight reads contribute 0)
  zero_ws_kernel<<<2048, 256, 0, stream>>>((unsigned int*)d_ws, (long)(off / 4));
  // 2. embedding gather -> bf16
  gather_kernel<<<SB, 64, 0, stream>>>(tok, X, x0);

  dim3 gP(G3P / 32, 16);   // input-projection grids: 108 N-tiles x 16 M-groups
  // layer 0
  wmma_gemm_kernel<<<gP, 256, 0, stream>>>(x0, DP, w_ih0, DD, DP, G3, b_ih0,
                                           xp, G3P, G3P, 0);
  gru_recur_kernel<<<RBLK, 32, 0, stream>>>(xp, w_hh0, b_hh0, h, hp, out0, cnt + 0);
  // layer 1
  wmma_gemm_kernel<<<gP, 256, 0, stream>>>(out0, HP, w_ih1, HH, HP, G3, b_ih1,
                                           xp, G3P, G3P, 0);
  gru_recur_kernel<<<RBLK, 32, 0, stream>>>(xp, w_hh1, b_hh1, h, hp, out1, cnt + 1);
  // layer 2
  wmma_gemm_kernel<<<gP, 256, 0, stream>>>(out1, HP, w_ih2, HH, HP, G3, b_ih2,
                                           xp, G3P, G3P, 0);
  gru_recur_kernel<<<RBLK, 32, 0, stream>>>(xp, w_hh2, b_hh2, h, hp, out2, cnt + 2);
  // sense layer (input = layer-0 output)
  wmma_gemm_kernel<<<gP, 256, 0, stream>>>(out0, HP, w_ihs, HH, HP, G3, b_ihs,
                                           xp, G3P, G3P, 0);
  gru_recur_kernel<<<RBLK, 32, 0, stream>>>(xp, w_hhs, b_hhs, h, hp, sout, cnt + 3);

  // output projections straight into d_out (logits), then in-place log_softmax
  wmma_gemm_kernel<<<dim3(NG / 32, 16), 256, 0, stream>>>(
      out2, HP, w_glob, HH, HP, NG, b_glob, outG, NG, NG, /*permute=*/1);
  wmma_gemm_kernel<<<dim3((NS + 31) / 32, 16), 256, 0, stream>>>(
      sout, HP, w_sens, HH, HP, NS, b_sens, outS, NS, NS, /*permute=*/0);
  logsoftmax_kernel<<<SB, 256, 0, stream>>>(outG, NG);
  logsoftmax_kernel<<<SB, 256, 0, stream>>>(outS, NS);
}
